// GraphAttention_62277025792037
// MI455X (gfx1250) — matrix operand order repaired and checked by reading, not compiled, not run on hardware
//
#include <hip/hip_runtime.h>
#include <hip/hip_fp16.h>

// Sizes from the reference
#define VV    10000   // vocab
#define EE    300     // embed dim
#define BB    32      // batch
#define LL    128     // seq len
#define KCH   313     // ceil(10000 / 32) K-chunks of the WMMA GEMM
#define NT2   20      // N-tiles of 16 (E=300 padded to 320; tile 19 is zeros)
#define NP2   (NT2 * 16)        // 320 padded N
#define SLABDW (NT2 * 256)      // 5120 dwords = one k-chunk's B slab (20 KB)
#define SLABBYTES (SLABDW * 4)

typedef __attribute__((ext_vector_type(16))) _Float16 v16h;
typedef __attribute__((ext_vector_type(8)))  float    v8f;

union ABfrag { unsigned u[8]; uint4 q[2]; v16h h; };

static __device__ __forceinline__ unsigned pack2h(float a, float b) {
  unsigned short lo = __half_as_ushort(__float2half(a));
  unsigned short hi = __half_as_ushort(__float2half(b));
  return (unsigned)lo | ((unsigned)hi << 16);
}

// ---------------------------------------------------------------------------
// Pack concept_embed_weight [V,300] f32 -> f16 in *fragment-major* layout:
//   cewp2[kc][nt][lane][j] , lane = half*16+lanem , j = 0..7
//   holds the f16 pair { cew[2*kpair, n] (lo) , cew[2*kpair+1, n] (hi) } with
//   kpair = kc*16 + j + 8*half (contiguous k order per VGPR j / lane-half; A build matches),
//   n = nt*16 + lanem.  Zero padding for kpair >= 5000 or n >= 300.
// Each lane's 8 B-fragment dwords are 32 contiguous bytes; one k-chunk slab
// (NT2 tiles) is one contiguous 20 KB block -> async-copyable as a memcpy.
// ---------------------------------------------------------------------------
__global__ __launch_bounds__(256) void pack_kernel(const float* __restrict__ cew,
                                                   unsigned* __restrict__ cewp2) {
  int o = blockIdx.x * 256 + threadIdx.x;
  if (o >= KCH * SLABDW) return;
  int kc  = o / SLABDW;
  int rem = o - kc * SLABDW;
  int nt  = rem >> 8;            // 0..19
  int e   = rem & 255;
  int lane  = e >> 3;
  int j     = e & 7;
  int half  = lane >> 4;
  int lanem = lane & 15;
  int kpair = kc * 16 + j + 8 * half;
  int n     = nt * 16 + lanem;
  int v0    = kpair * 2;
  float f0 = 0.0f, f1 = 0.0f;
  if (v0 < VV && n < EE) {
    f0 = cew[(size_t)v0 * EE + n];
    f1 = cew[(size_t)(v0 + 1) * EE + n];
  }
  cewp2[o] = pack2h(f0, f1);
}

// ---------------------------------------------------------------------------
// ctx[e] = mean_b ( sum_{l<len_b} embed[clist[b,l], e] / len_b ), plus ||ctx||
// ---------------------------------------------------------------------------
__global__ __launch_bounds__(320) void ctx_kernel(const int* __restrict__ clist,
                                                  const int* __restrict__ clen,
                                                  const float* __restrict__ embed,
                                                  float* __restrict__ ctx,
                                                  float* __restrict__ ctxn) {
  __shared__ float red[320];
  int e = threadIdx.x;
  float acc = 0.0f;
  if (e < EE) {
    for (int b = 0; b < BB; ++b) {
      int len = clen[b];
      float rs = 0.0f;
      for (int l = 0; l < len; ++l) {
        int c = clist[b * LL + l];
        rs += embed[(size_t)c * EE + e];
      }
      acc += rs / (float)len;
    }
    acc *= (1.0f / (float)BB);
  }
  ctx[e] = (e < EE) ? acc : 0.0f;
  red[e] = (e < EE) ? acc * acc : 0.0f;
  __syncthreads();
  if (e < 64) red[e] += red[e + 256];
  __syncthreads();
  for (int off = 128; off >= 1; off >>= 1) {
    if (e < off) red[e] += red[e + off];
    __syncthreads();
  }
  if (e == 0) ctxn[0] = sqrtf(red[0]);
}

// ---------------------------------------------------------------------------
// cos[v] = |dot(cew[v], ctx)| / max(||cew[v]|| * ||ctx||, 1e-8).  One wave/row.
// ---------------------------------------------------------------------------
__global__ __launch_bounds__(256) void cos_kernel(const float* __restrict__ cew,
                                                  const float* __restrict__ ctx,
                                                  const float* __restrict__ ctxn,
                                                  float* __restrict__ cosout) {
  int lane = threadIdx.x & 31;
  int wid  = threadIdx.x >> 5;
  int v = blockIdx.x * 8 + wid;          // grid = 1250 -> v in [0,10000)
  const float* r = cew + (size_t)v * EE;
  float d = 0.0f, q = 0.0f;
  for (int i = lane; i < EE; i += 32) {
    float x = r[i];
    d += x * ctx[i];
    q += x * x;
  }
  for (int m = 16; m >= 1; m >>= 1) {
    d += __shfl_xor(d, m);
    q += __shfl_xor(q, m);
  }
  if (lane == 0) {
    float denom = sqrtf(q) * ctxn[0];
    cosout[v] = fabsf(d) / fmaxf(denom, 1e-8f);
  }
}

// logit(e, cos, aff) for lambda=0.5, conc=2: conc*(l*e*cos + (1-l)*(e>0)*aff)
static __device__ __forceinline__ float logitf(float e, float c, float a) {
  return e * c + ((e > 0.0f) ? a : 0.0f);
}

// ---------------------------------------------------------------------------
// Main fused kernel, one block = 16 rows (one batch index b), 4 waves:
//   pass1: row-max of logits      (float4 scans, 128B/row/iter coalesced)
//   pass2: row sum of exp(logit - max)
//   pass3: double-buffered K loop:
//     - GLOBAL_LOAD_ASYNC_TO_LDS_B128 stages the next k-chunk's B slab
//       (single base + offset:t*512 immediates; ASYNCcnt pipelining,
//        s_wait_asynccnt 0xa leaves only the prefetch outstanding)
//     - 128 threads build the 16x32 f16 attn tile in LDS (fragment-major)
//     - each wave: 2x ds_load_b128 A frag, 10x ds_load_b128 B frags (batched),
//       then 5 unconditional v_wmma_f32_16x16x32_f16 (EXEC all-ones)
//   epilogue: masked/len-weighted pooling of C/D fragments -> partial[tile,n]
// ---------------------------------------------------------------------------
__global__ __launch_bounds__(128) void graph_attn_main(
    const int* __restrict__ clist, const int* __restrict__ clen,
    const float* __restrict__ edge, const float* __restrict__ aff,
    const unsigned* __restrict__ cewp2, const float* __restrict__ cosv,
    float* __restrict__ partial) {
  __shared__ __align__(16) unsigned bLDS[2][SLABDW];   // 2 x 20 KB B slabs
  __shared__ __align__(16) unsigned aLDS[256];         // 16x32 attn tile
  __shared__ int   c_sh[16];
  __shared__ float wrow[16];
  __shared__ float m_sh[16];
  __shared__ float is_sh[16];
  __shared__ float red[128];

  const int tid  = threadIdx.x;
  const int tile = blockIdx.x;            // 0..255
  const int b     = tile >> 3;            // 8 tiles per batch row (L=128)
  const int lbase = (tile & 7) * 16;

  if (tid < 16) {
    int l = lbase + tid;
    c_sh[tid] = clist[b * LL + l];
    int len = clen[b];
    wrow[tid] = (l < len) ? (1.0f / (float)len) : 0.0f;   // ragged-mean weight
  }
  __syncthreads();

  const int row = tid >> 3;               // 16 rows x 8 scanners
  const int sub = tid & 7;
  const float* erow = edge + (size_t)c_sh[row] * VV;

  // ---- pass 1: row max of logits (logits >= 0, so init 0 is safe) ----
  float mx = 0.0f;
  for (int v = sub * 4; v < VV; v += 32) {              // VV % 4 == 0
    float4 e4 = *(const float4*)(erow + v);
    float4 c4 = *(const float4*)(cosv + v);
    float4 a4 = *(const float4*)(aff  + v);
    mx = fmaxf(mx, logitf(e4.x, c4.x, a4.x));
    mx = fmaxf(mx, logitf(e4.y, c4.y, a4.y));
    mx = fmaxf(mx, logitf(e4.z, c4.z, a4.z));
    mx = fmaxf(mx, logitf(e4.w, c4.w, a4.w));
  }
  red[tid] = mx;
  __syncthreads();
  if (sub == 0) {
    float m = red[tid];
    #pragma unroll
    for (int i = 1; i < 8; ++i) m = fmaxf(m, red[tid + i]);
    m_sh[row] = m;
  }
  __syncthreads();
  const float mrow = m_sh[row];

  // ---- pass 2: row sum of exp ----
  float sm = 0.0f;
  for (int v = sub * 4; v < VV; v += 32) {
    float4 e4 = *(const float4*)(erow + v);
    float4 c4 = *(const float4*)(cosv + v);
    float4 a4 = *(const float4*)(aff  + v);
    sm += __expf(logitf(e4.x, c4.x, a4.x) - mrow);
    sm += __expf(logitf(e4.y, c4.y, a4.y) - mrow);
    sm += __expf(logitf(e4.z, c4.z, a4.z) - mrow);
    sm += __expf(logitf(e4.w, c4.w, a4.w) - mrow);
  }
  red[tid] = sm;
  __syncthreads();
  if (sub == 0) {
    float s = 0.0f;
    #pragma unroll
    for (int i = 0; i < 8; ++i) s += red[tid + i];
    is_sh[row] = 1.0f / s;
  }
  __syncthreads();

  // ---- pass 3: async-staged, double-buffered WMMA GEMM ----
  const int lane  = tid & 31;
  const int wid   = tid >> 5;
  const int half  = (lane >> 4) & 1;
  const int lanem = lane & 15;

  const unsigned long long gbase = (unsigned long long)(size_t)cewp2;
  const unsigned lds0 = (unsigned)(size_t)&bLDS[0][0];   // low 32 bits = LDS addr
  const unsigned lds1 = (unsigned)(size_t)&bLDS[1][0];
  const unsigned laneoff = (unsigned)(wid * 10) * 512u + (unsigned)lane * 16u;

  // prologue: stage k-chunk 0 into buffer 0 (10 async b128 per wave; the
  // instruction offset applies to BOTH the LDS and global side of the copy)
  {
    unsigned long long g = gbase + laneoff;
    unsigned l = lds0 + laneoff;
    #pragma unroll
    for (int t = 0; t < 10; ++t) {
      asm volatile("global_load_async_to_lds_b128 %0, %1, off offset:%c2"
                   :: "v"(l), "v"(g), "n"(t * 512) : "memory");
    }
  }

  v8f acc[5];
  #pragma unroll
  for (int i = 0; i < 5; ++i)
    #pragma unroll
    for (int j = 0; j < 8; ++j) acc[i][j] = 0.0f;

  for (int kc = 0; kc < KCH; ++kc) {
    const int cur = kc & 1;
    __syncthreads();   // prior iteration's LDS reads complete before overwrite

    // prefetch next slab into the other buffer
    if (kc + 1 < KCH) {
      unsigned long long g = gbase + (unsigned long long)(kc + 1) * SLABBYTES + laneoff;
      unsigned l = (cur ? lds0 : lds1) + laneoff;
      #pragma unroll
      for (int t = 0; t < 10; ++t) {
        asm volatile("global_load_async_to_lds_b128 %0, %1, off offset:%c2"
                     :: "v"(l), "v"(g), "n"(t * 512) : "memory");
      }
    }

    // cooperative build of the 16x32 attn tile, fragment-major: aLDS[lane*8+j]
    #pragma unroll
    for (int rep = 0; rep < 2; ++rep) {
      int idx = tid + rep * 128;
      int ln  = idx >> 3;                 // destination lane 0..31
      int jj  = idx & 7;                  // fragment VGPR j
      int hf  = ln >> 4;
      int r   = ln & 15;                  // matrix row M
      int kp  = jj + 8 * hf;              // contiguous kpair (matches B pack)
      int v0  = kc * 32 + kp * 2;
      float p0 = 0.0f, p1 = 0.0f;
      if (v0 < VV) {                      // v0 even => v0+1 also < VV
        float2 e2 = *(const float2*)(edge + (size_t)c_sh[r] * VV + v0);
        float2 c2 = *(const float2*)(cosv + v0);
        float2 a2 = *(const float2*)(aff  + v0);
        float is = is_sh[r], m = m_sh[r];
        p0 = __expf(logitf(e2.x, c2.x, a2.x) - m) * is;
        p1 = __expf(logitf(e2.y, c2.y, a2.y) - m) * is;
      }
      aLDS[idx] = pack2h(p0, p1);
    }

    // wait for current slab (leave only the 10 prefetch ops outstanding)
    if (kc + 1 < KCH) asm volatile("s_wait_asynccnt 0xa" ::: "memory");
    else              asm volatile("s_wait_asynccnt 0x0" ::: "memory");
    __syncthreads();   // publish async LDS writes + attn tile to all waves

    // A fragment: 2 x ds_load_b128
    ABfrag a;
    const uint4* ap = (const uint4*)&aLDS[lane * 8];
    a.q[0] = ap[0];
    a.q[1] = ap[1];

    // batch all 5 B fragments (10 x ds_load_b128), then 5 unconditional WMMAs
    ABfrag bf[5];
    #pragma unroll
    for (int i = 0; i < 5; ++i) {
      int nt = wid * 5 + i;               // N-tiles wid*5..wid*5+4; tile 19 = pad
      const uint4* bp = (const uint4*)&bLDS[cur][nt * 256 + lane * 8];
      bf[i].q[0] = bp[0];
      bf[i].q[1] = bp[1];
    }
    #pragma unroll
    for (int i = 0; i < 5; ++i) {
      acc[i] = __builtin_amdgcn_wmma_f32_16x16x32_f16(
          false, a.h, false, bf[i].h, (short)0, acc[i], false, false);
    }
  }

  // ---- masked/len-weighted pooling of C/D fragment: M = j + 8*half ----
  #pragma unroll
  for (int i = 0; i < 5; ++i) {
    int nt = wid * 5 + i;
    float t = 0.0f;
    #pragma unroll
    for (int j = 0; j < 8; ++j) t += acc[i][j] * wrow[j + 8 * half];
    t += __shfl_xor(t, 16);               // combine the two M-halves (same N)
    if (half == 0) partial[(size_t)tile * NP2 + nt * 16 + lanem] = t;
  }
}

// ---------------------------------------------------------------------------
// out[b, n] = sum over the 8 tiles of batch row b (deterministic, no atomics)
// ---------------------------------------------------------------------------
__global__ __launch_bounds__(320) void finalize_kernel(const float* __restrict__ partial,
                                                       float* __restrict__ out) {
  int bq = blockIdx.x;
  int n  = threadIdx.x;
  if (n < EE) {
    float s = 0.0f;
    #pragma unroll
    for (int j = 0; j < 8; ++j) s += partial[(size_t)(bq * 8 + j) * NP2 + n];
    out[bq * EE + n] = s;
  }
}

extern "C" void kernel_launch(void* const* d_in, const int* in_sizes, int n_in,
                              void* d_out, int out_size, void* d_ws, size_t ws_size,
                              hipStream_t stream) {
  const int*   clist = (const int*)d_in[0];     // [B,L]
  const int*   clen  = (const int*)d_in[1];     // [B]
  const float* edge  = (const float*)d_in[2];   // [V,V]
  const float* aff   = (const float*)d_in[3];   // [V]
  const float* cew   = (const float*)d_in[4];   // [V,E]
  const float* embed = (const float*)d_in[5];   // [V,E]
  float*       out   = (float*)d_out;           // [B,E]

  // workspace layout (dwords)
  unsigned* cewp2   = (unsigned*)d_ws;                        // KCH*SLABDW
  float*    cosv    = (float*)d_ws + (size_t)KCH * SLABDW;    // V
  float*    ctx     = cosv + VV;                              // 320
  float*    ctxn    = ctx + 320;                              // 1
  float*    partial = ctxn + 1;                               // 256*NP2

  {
    int entries = KCH * SLABDW;
    int blocks  = (entries + 255) / 256;
    pack_kernel<<<blocks, 256, 0, stream>>>(cew, cewp2);
  }
  ctx_kernel<<<1, 320, 0, stream>>>(clist, clen, embed, ctx, ctxn);
  cos_kernel<<<VV / 8, 256, 0, stream>>>(cew, ctx, ctxn, cosv);
  graph_attn_main<<<(BB * LL) / 16, 128, 0, stream>>>(clist, clen, edge, aff,
                                                      cewp2, cosv, partial);
  finalize_kernel<<<BB, 320, 0, stream>>>(partial, out);
}
